// SCAM_39917426049705
// MI455X (gfx1250) — matrix operand order repaired and checked by reading, not compiled, not run on hardware
//
#include <hip/hip_runtime.h>

// ---------------------------------------------------------------------------
// SCAM stereo cross-attention for MI455X (gfx1250, wave32, WMMA).
// All big contractions use v_wmma_f32_16x16x32_f16 (f16 in, f32 accum).
// Window attention is a flash-style streaming softmax (no P materialized).
// All WMMA fragments are loaded as aligned 128-bit chunks (ds/global b128).
// Relative-position bias is pre-gathered to a 39x39 LDS table per block.
// Softmax cross-lane reductions use VALU DPP16 (no LDS/bpermute traffic).
// ---------------------------------------------------------------------------

typedef _Float16 h16;
typedef __attribute__((ext_vector_type(8)))  _Float16 v8h;
typedef __attribute__((ext_vector_type(16))) _Float16 v16h;
typedef __attribute__((ext_vector_type(8)))  float    v8f;

#define R_TOT   18432          // B * S * S = 2*96*96
#define S_DIM   96
#define C_DIM   180
#define C_PAD   192
#define NHEAD   6
#define DHEAD   30
#define SCALE_F 0.18257418583505536f   // 30^-0.5
#define LN_EPS  1e-6f

// ---- WMMA fragment helpers (CDNA5 16x16x32 f16 layouts, ISA 7.12.2) -------
// A (16x32, 16-bit): lane m=lane&15, hi=lane>>4; elems e: K=((e<8)?e:e+8)+8*hi
//   -> two contiguous 8-half (16B) chunks at k0+8*hi and k0+16+8*hi.
// B (32x16, 16-bit): lane n=lane&15, hi=lane>>4; elems e: K=((e<8)?e:e+8)+8*hi
//   -> two contiguous 8-half (16B) chunks at k0+8*hi and k0+16+8*hi.
// C/D (16x16 f32):   elem r -> (M = r + 8*hi, N = lane&15)

static __device__ __forceinline__ v16h cat8(v8h lo, v8h hi8) {
  return __builtin_shufflevector(lo, hi8, 0, 1, 2, 3, 4, 5, 6, 7,
                                          8, 9, 10, 11, 12, 13, 14, 15);
}

static __device__ __forceinline__ v16h load_a_frag(const h16* base, int row, int ld, int k0, int hi) {
  const h16* p = base + (size_t)row * (size_t)ld + k0;
  const v8h lo  = *(const v8h*)(p + 8 * hi);       // 16B aligned
  const v8h hi8 = *(const v8h*)(p + 16 + 8 * hi);  // 16B aligned
  return cat8(lo, hi8);
}

// B^T stored row-major [n][k]: two 8-half chunks per lane (same K order as A).
static __device__ __forceinline__ v16h load_b_frag(const h16* base, int rowN, int ld, int k0, int hi) {
  const h16* p = base + (size_t)rowN * (size_t)ld + k0;
  const v8h lo  = *(const v8h*)(p + 8 * hi);       // 16B aligned
  const v8h hi8 = *(const v8h*)(p + 16 + 8 * hi);
  return cat8(lo, hi8);
}

static __device__ __forceinline__ v8f wmma16(v16h a, v16h b, v8f c) {
  return __builtin_amdgcn_wmma_f32_16x16x32_f16(false, a, false, b, (short)0, c, false, false);
}

// ---- 16-lane butterfly reductions on the VALU via DPP16 -------------------
// quad_perm(0xB1): i^1; quad_perm(0x4E): i^2; row_half_mirror: i<->7-i in 8;
// row_mirror: i<->15-i in 16. Four involutions => all 16 lanes hold result,
// and nothing crosses the 16-lane half boundary (halves own different rows).
template <int CTRL>
static __device__ __forceinline__ float dpp_f(float v) {
  return __int_as_float(__builtin_amdgcn_update_dpp(
      0, __float_as_int(v), CTRL, 0xF, 0xF, true));
}
static __device__ __forceinline__ float red16_max(float v) {
  v = fmaxf(v, dpp_f<0x0B1>(v));
  v = fmaxf(v, dpp_f<0x04E>(v));
  v = fmaxf(v, dpp_f<0x141>(v));
  v = fmaxf(v, dpp_f<0x140>(v));
  return v;
}
static __device__ __forceinline__ float red16_sum(float v) {
  v += dpp_f<0x0B1>(v);
  v += dpp_f<0x04E>(v);
  v += dpp_f<0x141>(v);
  v += dpp_f<0x140>(v);
  return v;
}

// ---------------------------------------------------------------------------
// Kernel 1: per-row LayerNorm over C=180 (or plain cast), f32 -> f16 padded
// to stride 192 with zeros (so WMMA K-steps past c=179 contribute 0).
// ---------------------------------------------------------------------------
__global__ void __launch_bounds__(256)
rownorm_kernel(const float* __restrict__ X, const float* __restrict__ w,
               const float* __restrict__ bvec, h16* __restrict__ Y, int do_norm) {
  __shared__ float sm[256];
  const int row = blockIdx.x, t = threadIdx.x;
  float x = (t < C_DIM) ? X[(size_t)row * C_DIM + t] : 0.f;
  float val = x;
  if (do_norm) {
    sm[t] = x; __syncthreads();
#pragma unroll
    for (int s = 128; s > 0; s >>= 1) { if (t < s) sm[t] += sm[t + s]; __syncthreads(); }
    const float mean = sm[0] * (1.f / C_DIM);
    __syncthreads();
    const float d = (t < C_DIM) ? (x - mean) : 0.f;
    sm[t] = d * d; __syncthreads();
#pragma unroll
    for (int s = 128; s > 0; s >>= 1) { if (t < s) sm[t] += sm[t + s]; __syncthreads(); }
    const float var = sm[0] * (1.f / C_DIM);
    const int tc = (t < C_DIM) ? t : 0;
    val = d / sqrtf(var + LN_EPS) * w[tc] + bvec[tc];
  }
  if (t < C_PAD) Y[(size_t)row * C_PAD + t] = (t < C_DIM) ? (h16)val : (h16)0.f;
}

// ---------------------------------------------------------------------------
// Kernel 2: cast 180x180 f32 weight to 192x192 f16 (zero padded)
// ---------------------------------------------------------------------------
__global__ void __launch_bounds__(256)
castw_kernel(const float* __restrict__ W, h16* __restrict__ Wo) {
  int idx = blockIdx.x * 256 + threadIdx.x;
  if (idx >= C_PAD * C_PAD) return;
  int o = idx / C_PAD, c = idx % C_PAD;
  Wo[idx] = (o < C_DIM && c < C_DIM) ? (h16)W[o * C_DIM + c] : (h16)0.f;
}

// ---------------------------------------------------------------------------
// Kernel 3: dual-direction windowed attention, flash-streaming softmax.
// grid = 2 dir * 6 heads * 72 windows = 864 blocks of 256 threads (8 waves).
// Wave owns 32 queries; 18 key-blocks of 32; out = P @ K (keys are values).
// LDS: Q 16KB + K 36KB + K^T 36KB + P 8KB + bias 6KB = 102KB (of 320KB/WG).
// ---------------------------------------------------------------------------
__global__ void __launch_bounds__(256)
winattn_kernel(const h16* __restrict__ nl, const h16* __restrict__ nr,
               const float* __restrict__ rpb,
               float* __restrict__ xl_o, float* __restrict__ xr_o) {
  __shared__ __attribute__((aligned(16))) h16 Qs[256 * 32];        // queries, d pad 30->32
  __shared__ __attribute__((aligned(16))) h16 Ks[576 * 32];        // keys row-major
  __shared__ __attribute__((aligned(16))) h16 KT[32 * 576];        // keys transposed
  __shared__ __attribute__((aligned(16))) h16 Pb_all[8 * 16 * 32]; // per-wave P tile
  __shared__ float Bt[39 * 39];                                    // rel-pos bias table

  const int bx = blockIdx.x;
  const int dir = bx / 432;
  int rem = bx % 432;
  const int head = rem / 72;
  const int w = rem % 72;
  const int b = w / 36, wi = w % 36, wr = wi / 6, wc = wi % 6;
  const h16* qsrc = dir ? nr : nl;
  const h16* ksrc = dir ? nl : nr;
  float* outp = dir ? xr_o : xl_o;
  const int tid = threadIdx.x;

  // bias table: Bt[(dr+15)*39 + (dc+15)] = rpb[mod(39*(dr-7)+(dc-7),1521)*6+h]
  for (int idx = tid; idx < 1521; idx += 256) {
    const int a = idx / 39, bc = idx % 39;          // dr+15, dc+15
    int t = 39 * (a - 22) + (bc - 22);              // 39*(dr-7) + (dc-7)
    if (t < 0) t += 1521;
    Bt[idx] = rpb[t * NHEAD + head];
  }
  { // load Q window (16x16 queries, head slice, pad d to 32)
    const int q = tid, qi = q >> 4, qj = q & 15;
    const size_t grow = (size_t)b * 9216 + (size_t)(wr * 16 + qi) * S_DIM + (wc * 16 + qj);
    const h16* src = qsrc + grow * C_PAD + head * DHEAD;
#pragma unroll
    for (int f = 0; f < DHEAD; ++f) Qs[q * 32 + f] = src[f];
    Qs[q * 32 + 30] = (h16)0.f; Qs[q * 32 + 31] = (h16)0.f;
  }
  // load K window (24x24 keys from padded image; OOB -> exact zeros like ref)
  for (int k = tid; k < 576; k += 256) {
    const int ki = k / 24, kj = k % 24;
    const int hh = wr * 16 + ki - 4, ww = wc * 16 + kj - 4;
    const bool valid = (hh >= 0 && hh < S_DIM && ww >= 0 && ww < S_DIM);
    const h16* src = ksrc + ((size_t)b * 9216 + (size_t)(hh * S_DIM + ww)) * C_PAD + head * DHEAD;
#pragma unroll
    for (int f = 0; f < DHEAD; ++f) {
      const h16 v = valid ? src[f] : (h16)0.f;
      Ks[k * 32 + f] = v;
      KT[f * 576 + k] = v;
    }
    Ks[k * 32 + 30] = (h16)0.f; Ks[k * 32 + 31] = (h16)0.f;
    KT[30 * 576 + k] = (h16)0.f; KT[31 * 576 + k] = (h16)0.f;
  }
  __syncthreads();

  const int wid = tid >> 5, lane = tid & 31, lanen = lane & 15, hi = lane >> 4;
  const int qbase = wid * 32;
  h16* Pb = Pb_all + wid * 16 * 32;

  v16h aq[2];
  aq[0] = load_a_frag(Qs, qbase + lanen, 32, 0, hi);
  aq[1] = load_a_frag(Qs, qbase + 16 + lanen, 32, 0, hi);

  // hoisted per-query bias offsets: qoff = 39*qi + qj
  int qoff[2][8];
#pragma unroll
  for (int mt = 0; mt < 2; ++mt)
#pragma unroll
    for (int r = 0; r < 8; ++r) {
      const int q = qbase + mt * 16 + hi * 8 + r;
      qoff[mt][r] = 39 * (q >> 4) + (q & 15);
    }

  float mrow[2][8], lrow[2][8];
  v8f acc[2][2];
#pragma unroll
  for (int mt = 0; mt < 2; ++mt) {
    acc[mt][0] = {}; acc[mt][1] = {};
#pragma unroll
    for (int r = 0; r < 8; ++r) { mrow[mt][r] = -1e30f; lrow[mt][r] = 0.f; }
  }

  for (int kb = 0; kb < 18; ++kb) {
    const v16h bs0 = load_b_frag(Ks, kb * 32 + lanen, 32, 0, hi);
    const v16h bs1 = load_b_frag(Ks, kb * 32 + 16 + lanen, 32, 0, hi);
    const int key0 = kb * 32 + lanen, key1 = key0 + 16;
    const int ki0 = key0 / 24, kj0 = key0 - ki0 * 24;
    const int ki1 = key1 / 24, kj1 = key1 - ki1 * 24;
    const int kk0 = 39 * ki0 + kj0 + 600;   // +15*39+15; idx = kk - qoff in [0,1520]
    const int kk1 = 39 * ki1 + kj1 + 600;
#pragma unroll
    for (int mt = 0; mt < 2; ++mt) {
      const v8f s0 = wmma16(aq[mt], bs0, (v8f){});
      const v8f s1 = wmma16(aq[mt], bs1, (v8f){});
#pragma unroll
      for (int r = 0; r < 8; ++r) {
        const float sc0 = s0[r] * SCALE_F + Bt[kk0 - qoff[mt][r]];
        const float sc1 = s1[r] * SCALE_F + Bt[kk1 - qoff[mt][r]];
        const float bm = red16_max(fmaxf(sc0, sc1));
        const float mn = fmaxf(mrow[mt][r], bm);
        const float corr = __expf(mrow[mt][r] - mn);
        const float p0 = __expf(sc0 - mn);
        const float p1 = __expf(sc1 - mn);
        const float bsum = red16_sum(p0 + p1);
        lrow[mt][r] = lrow[mt][r] * corr + bsum;
        mrow[mt][r] = mn;
        acc[mt][0][r] *= corr;
        acc[mt][1][r] *= corr;
        Pb[(hi * 8 + r) * 32 + lanen]      = (h16)p0;
        Pb[(hi * 8 + r) * 32 + 16 + lanen] = (h16)p1;
      }
      const v16h ap = load_a_frag(Pb, lanen, 32, 0, hi);
#pragma unroll
      for (int ft = 0; ft < 2; ++ft) {
        const v16h bk = load_b_frag(KT, 16 * ft + lanen, 576, kb * 32, hi);
        acc[mt][ft] = wmma16(ap, bk, acc[mt][ft]);
      }
    }
  }
  // epilogue: out[q, f] = acc / l   -> scatter to (B,S,S,C) f32
#pragma unroll
  for (int mt = 0; mt < 2; ++mt)
#pragma unroll
    for (int ft = 0; ft < 2; ++ft) {
      const int f = 16 * ft + lanen;
      if (f < DHEAD) {
#pragma unroll
        for (int r = 0; r < 8; ++r) {
          const int q = qbase + mt * 16 + hi * 8 + r;
          const int qi = q >> 4, qj = q & 15;
          const size_t grow = (size_t)b * 9216 + (size_t)(wr * 16 + qi) * S_DIM + (wc * 16 + qj);
          outp[grow * C_DIM + head * DHEAD + f] = acc[mt][ft][r] / lrow[mt][r];
        }
      }
    }
}

// ---------------------------------------------------------------------------
// Kernel 4: Y(rows x 192,f16) = X(rows x 192,f16) @ W^T + bias.
// 128-row tile in LDS; W (padded f16) streamed from global (L2 resident).
// ---------------------------------------------------------------------------
__global__ void __launch_bounds__(256)
gemm180_kernel(const h16* __restrict__ X, const h16* __restrict__ Wg,
               const float* __restrict__ bias, h16* __restrict__ Y) {
  __shared__ __attribute__((aligned(16))) h16 Xs[128 * C_PAD];   // 48 KB
  const int rb = blockIdx.x, tid = threadIdx.x;
  { // vectorized 16B staging copy
    const uint4* src = (const uint4*)(X + (size_t)rb * 128 * C_PAD);
    uint4* dst = (uint4*)Xs;
#pragma unroll
    for (int idx = tid; idx < 128 * C_PAD / 8; idx += 256) dst[idx] = src[idx];
  }
  __syncthreads();
  const int wid = tid >> 5, lane = tid & 31, lanen = lane & 15, hi = lane >> 4;
  v16h a[6];
#pragma unroll
  for (int ks = 0; ks < 6; ++ks) a[ks] = load_a_frag(Xs, wid * 16 + lanen, C_PAD, ks * 32, hi);
  for (int nt = 0; nt < 12; ++nt) {
    v8f acc = {};
#pragma unroll
    for (int ks = 0; ks < 6; ++ks) {
      const v16h bf = load_b_frag(Wg, nt * 16 + lanen, C_PAD, ks * 32, hi);
      acc = wmma16(a[ks], bf, acc);
    }
    const int n = nt * 16 + lanen;
    const float bn = (n < C_DIM) ? bias[n] : 0.f;
#pragma unroll
    for (int r = 0; r < 8; ++r) {
      const int m = wid * 16 + hi * 8 + r;
      const size_t grow = (size_t)rb * 128 + m;
      const float v = (n < C_DIM) ? (acc[r] + bn) : 0.f;
      Y[grow * C_PAD + n] = (h16)v;
    }
  }
}

// ---------------------------------------------------------------------------
// Kernel 5: cross-attention scores S[j,k] = SCALE * <A_l[b,j,i,:], A_r[b,k,i,:]>
// per (b,i). Fragments read straight from global (contiguous in padded rows).
// ---------------------------------------------------------------------------
__global__ void __launch_bounds__(256)
cross_scores_kernel(const h16* __restrict__ Al, const h16* __restrict__ Ar,
                    float* __restrict__ Sws) {
  const int bx = blockIdx.x;               // 192 = b*96 + i
  const int b = bx / S_DIM, i = bx % S_DIM;
  const int tid = threadIdx.x, wid = tid >> 5, lane = tid & 31, lanen = lane & 15, hi = lane >> 4;
  const h16* Abase = Al + ((size_t)b * 9216 + i) * C_PAD;  // + j*96*192
  const h16* Bbase = Ar + ((size_t)b * 9216 + i) * C_PAD;
  float* Sb = Sws + (size_t)bx * 9216;
  const int ldg = S_DIM * C_PAD;           // 18432 halves between rows j
  for (int job = wid; job < 36; job += 8) {
    const int mt = job / 6, nt = job % 6;
    v8f acc = {};
#pragma unroll
    for (int ks = 0; ks < 6; ++ks) {
      const v16h a  = load_a_frag(Abase, mt * 16 + lanen, ldg, ks * 32, hi);
      const v16h bf = load_b_frag(Bbase, nt * 16 + lanen, ldg, ks * 32, hi);
      acc = wmma16(a, bf, acc);
    }
#pragma unroll
    for (int r = 0; r < 8; ++r)
      Sb[(mt * 16 + hi * 8 + r) * S_DIM + nt * 16 + lanen] = acc[r] * SCALE_F;
  }
}

// ---------------------------------------------------------------------------
// Kernel 6: softmax (row for side0 / col-transposed for side1) + P @ V,
// residual epilogue out = feat + coef * F.  grid = 2 sides * 192.
// ---------------------------------------------------------------------------
__global__ void __launch_bounds__(256)
cross_out_kernel(const float* __restrict__ Sws,
                 const h16* __restrict__ Vl, const h16* __restrict__ Vr,
                 const float* __restrict__ feat_l, const float* __restrict__ feat_r,
                 const float* __restrict__ beta, const float* __restrict__ gamma,
                 float* __restrict__ out) {
  __shared__ __attribute__((aligned(16))) h16 Ps[S_DIM * S_DIM]; // probabilities (A)
  __shared__ __attribute__((aligned(16))) h16 VT[C_PAD * S_DIM]; // V^T (B contiguous)
  const int bx = blockIdx.x;
  const int side = bx / 192;
  const int idx = bx % 192;
  const int b = idx / S_DIM, i = idx % S_DIM;
  const int tid = threadIdx.x;
  const h16* Vsrc = side ? Vl : Vr;
  const float* feat = side ? feat_r : feat_l;
  const float* coef = side ? gamma : beta;
  float* outp = out + (size_t)side * (size_t)R_TOT * C_DIM;
  const float* Sb = Sws + ((size_t)b * S_DIM + i) * 9216;

  for (int k = tid; k < S_DIM * C_PAD; k += 256) {
    const int t = k / C_PAD, c = k % C_PAD;
    VT[c * S_DIM + t] = Vsrc[((size_t)b * 9216 + (size_t)i * S_DIM + t) * C_PAD + c];
  }
  if (tid < S_DIM) {
    float mx = -1e30f, s = 0.f;
    if (side == 0) {  // softmax over k (last axis); Ps[j][k]
      for (int k = 0; k < S_DIM; ++k) mx = fmaxf(mx, Sb[tid * S_DIM + k]);
      for (int k = 0; k < S_DIM; ++k) s += __expf(Sb[tid * S_DIM + k] - mx);
      const float inv = 1.f / s;
      for (int k = 0; k < S_DIM; ++k)
        Ps[tid * S_DIM + k] = (h16)(__expf(Sb[tid * S_DIM + k] - mx) * inv);
    } else {          // softmax over j (axis -2), stored transposed: Ps[k][j]
      for (int j = 0; j < S_DIM; ++j) mx = fmaxf(mx, Sb[j * S_DIM + tid]);
      for (int j = 0; j < S_DIM; ++j) s += __expf(Sb[j * S_DIM + tid] - mx);
      const float inv = 1.f / s;
      for (int j = 0; j < S_DIM; ++j)
        Ps[tid * S_DIM + j] = (h16)(__expf(Sb[j * S_DIM + tid] - mx) * inv);
    }
  }
  __syncthreads();

  const int wid = tid >> 5, lane = tid & 31, lanen = lane & 15, hi = lane >> 4;
  for (int job = wid; job < 72; job += 8) {
    const int mt = job / 12, nt = job % 12;
    v8f acc = {};
#pragma unroll
    for (int ks = 0; ks < 3; ++ks) {
      const v16h a  = load_a_frag(Ps, mt * 16 + lanen, S_DIM, ks * 32, hi);
      const v16h bf = load_b_frag(VT, nt * 16 + lanen, S_DIM, ks * 32, hi);
      acc = wmma16(a, bf, acc);
    }
    const int c = nt * 16 + lanen;
    if (c < C_DIM) {
      const float cf = coef[c];
#pragma unroll
      for (int r = 0; r < 8; ++r) {
        const int m = mt * 16 + hi * 8 + r;
        const size_t grow = (size_t)b * 9216 + (size_t)i * S_DIM + m;
        outp[grow * C_DIM + c] = feat[grow * C_DIM + c] + cf * acc[r];
      }
    }
  }
}

// ---------------------------------------------------------------------------
extern "C" void kernel_launch(void* const* d_in, const int* in_sizes, int n_in,
                              void* d_out, int out_size, void* d_ws, size_t ws_size,
                              hipStream_t stream) {
  const float* feat_l = (const float*)d_in[0];
  const float* feat_r = (const float*)d_in[1];
  const float* rpb    = (const float*)d_in[2];
  const float* ln_l_w = (const float*)d_in[3];
  const float* ln_l_b = (const float*)d_in[4];
  const float* ln_r_w = (const float*)d_in[5];
  const float* ln_r_b = (const float*)d_in[6];
  const float* wl1 = (const float*)d_in[7];
  const float* bl1 = (const float*)d_in[8];
  const float* wr1 = (const float*)d_in[9];
  const float* br1 = (const float*)d_in[10];
  const float* wl2 = (const float*)d_in[11];
  const float* bl2 = (const float*)d_in[12];
  const float* wr2 = (const float*)d_in[13];
  const float* br2 = (const float*)d_in[14];
  const float* beta  = (const float*)d_in[15];
  const float* gamma = (const float*)d_in[16];
  float* out = (float*)d_out;

  // workspace carve-out (~66 MB total)
  const size_t F16B = (size_t)R_TOT * C_PAD * sizeof(h16);   // 7,077,888
  const size_t F32B = (size_t)R_TOT * C_DIM * sizeof(float); // 13,271,040
  char* base = (char*)d_ws;
  h16*   nl16   = (h16*)base;                       // LN rows (later: LN(x_l_))
  h16*   nr16   = (h16*)(base + F16B);              // LN rows (later: LN(x_r_))
  char*  regC   = base + 2 * F16B;                  // F32B region
  char*  regD   = regC + F32B;                      // F32B region
  float* xl_att = (float*)regC;                     // window-attn out L
  float* xr_att = (float*)regD;                     // window-attn out R
  h16*   fl16   = (h16*)regC;                       // later: f16 cast of feat_l
  h16*   fr16   = (h16*)regD;                       // later: f16 cast of feat_r
  float* Sws    = (float*)regC;                     // later: cross scores (7MB)
  h16*   Al16   = (h16*)(regD + F32B);
  h16*   Ar16   = Al16 + (size_t)R_TOT * C_PAD;
  h16*   Vl16   = Ar16 + (size_t)R_TOT * C_PAD;
  h16*   Vr16   = Vl16 + (size_t)R_TOT * C_PAD;
  h16*   w1l    = Vr16 + (size_t)R_TOT * C_PAD;
  h16*   w1r    = w1l + C_PAD * C_PAD;
  h16*   w2l    = w1r + C_PAD * C_PAD;
  h16*   w2r    = w2l + C_PAD * C_PAD;

  // 1) pre-attention LayerNorms (reference uses ln_l for BOTH sides here)
  rownorm_kernel<<<R_TOT, 256, 0, stream>>>(feat_l, ln_l_w, ln_l_b, nl16, 1);
  rownorm_kernel<<<R_TOT, 256, 0, stream>>>(feat_r, ln_l_w, ln_l_b, nr16, 1);
  // weight casts (independent)
  castw_kernel<<<144, 256, 0, stream>>>(wl1, w1l);
  castw_kernel<<<144, 256, 0, stream>>>(wr1, w1r);
  castw_kernel<<<144, 256, 0, stream>>>(wl2, w2l);
  castw_kernel<<<144, 256, 0, stream>>>(wr2, w2r);
  // 2) both windowed attentions fused in one launch
  winattn_kernel<<<864, 256, 0, stream>>>(nl16, nr16, rpb, xl_att, xr_att);
  // 3) LN of attention outputs (reuse nl16/nr16 slots), f16 casts of feats
  rownorm_kernel<<<R_TOT, 256, 0, stream>>>(xl_att, ln_l_w, ln_l_b, nl16, 1);
  rownorm_kernel<<<R_TOT, 256, 0, stream>>>(xr_att, ln_r_w, ln_r_b, nr16, 1);
  rownorm_kernel<<<R_TOT, 256, 0, stream>>>(feat_l, ln_l_w, ln_l_b, fl16, 0);
  rownorm_kernel<<<R_TOT, 256, 0, stream>>>(feat_r, ln_l_w, ln_l_b, fr16, 0);
  // 4) four 1x1 convs as WMMA GEMMs
  gemm180_kernel<<<144, 256, 0, stream>>>(nl16, w1l, bl1, Al16);
  gemm180_kernel<<<144, 256, 0, stream>>>(nr16, w1r, br1, Ar16);
  gemm180_kernel<<<144, 256, 0, stream>>>(fl16, w2l, bl2, Vl16);
  gemm180_kernel<<<144, 256, 0, stream>>>(fr16, w2r, br2, Vr16);
  // 5) cross-attention scores (shared by both softmax directions)
  cross_scores_kernel<<<192, 256, 0, stream>>>(Al16, Ar16, Sws);
  // 6) softmax + P@V + residual for both outputs
  cross_out_kernel<<<384, 256, 0, stream>>>(Sws, Vl16, Vr16, feat_l, feat_r,
                                            beta, gamma, out);
}
